// PLSTMCell_89163521065379
// MI455X (gfx1250) — matrix operand order repaired and checked by reading, not compiled, not run on hardware
//
#include <hip/hip_runtime.h>
#include <math.h>

typedef __attribute__((ext_vector_type(16))) __bf16 v16bf;
typedef __attribute__((ext_vector_type(2)))  __bf16 v2bf;
typedef __attribute__((ext_vector_type(8)))  float  v8f;

#define B_    128
#define T_    2048
#define D_    256
#define U_    512
#define G4U   2048          // 4*U
#define KTOT  768           // D + U (fused [x_t | h] GEMM)
#define NKT   24            // KTOT/32 k-tiles
#define BT    16            // batch rows per WG (one WMMA M-tile)
#define US    32            // units per WG
#define NCOLS 128           // z columns per WG (4 gates * US)
#define NWG   128           // 8 batch tiles * 16 unit slices
#define THREADS 128         // 4 wave32
#define ALPHA 0.001f

// LDS partition (dynamic shared memory; total 229,376 B < 320 KB/WGP)
#define BFRAG_OFF 0         // 8*24 tiles * 512 halves  = 196608 B (resident weights, bf16)
#define AFRAG_OFF 196608    // 24 tiles * 512 halves    =  24576 B (A restaged per step)
#define Z_OFF     221184    // 128 cols * 16 rows f32   =   8192 B (column-major)
#define SMEM_BYTES 229376

// CDNA5 hardware tanh (TRANS32). One V_NOP after a TRANS op before its output
// is consumed (ISA 7.3 multicycle co-execution hazard); compiler can't see
// through the asm, so we pay it explicitly.
__device__ __forceinline__ float htanh(float v) {
#if defined(__AMDGCN__)
    float r;
    asm volatile("v_tanh_f32 %0, %1\n\tv_nop" : "=v"(r) : "v"(v));
    return r;
#else
    return tanhf(v);
#endif
}

__global__ void plstm_init(const float* __restrict__ h0,
                           float* __restrict__ hbuf,
                           int* __restrict__ bar) {
    int i = blockIdx.x * blockDim.x + threadIdx.x;
    if (i == 0) *bar = 0;
    if (i < B_ * U_) {
        hbuf[i] = h0[i];              // ping buffer = h0
        hbuf[B_ * U_ + i] = 0.0f;     // pong buffer (overwritten before read)
    }
}

__global__ __launch_bounds__(THREADS)
void plstm_persistent(const float* __restrict__ x,
                      const float* __restrict__ c0,
                      const float* __restrict__ wk,     // [D, 4U]
                      const float* __restrict__ wr,     // [U, 4U]
                      const float* __restrict__ bias,   // [4U]
                      const float* __restrict__ tg,     // [3, U]
                      float* __restrict__ out,          // [B, T, U]
                      float* __restrict__ hbuf,         // [2][B][U]
                      int* __restrict__ bar) {
    extern __shared__ char smem[];
    __bf16* Bfrag = (__bf16*)(smem + BFRAG_OFF);
    __bf16* Afrag = (__bf16*)(smem + AFRAG_OFF);
    float*  zsh   = (float*) (smem + Z_OFF);   // zsh[col*16 + row]

    const int tid   = threadIdx.x;
    const int lane  = tid & 31;
    const int wv    = tid >> 5;
    const int slice = blockIdx.x & 15;   // unit slice (16 of them)
    const int btile = blockIdx.x >> 4;   // batch tile (8 of them)
    const int b0 = btile * BT;
    const int u0 = slice * US;

    // ---- one-time: gather this WG's bf16 weight slice into LDS (fragment order) ----
    // Wcat[k, col]: k<256 from kernel, else recurrent; local col j in 0..127:
    // gate g=j/32, unit u0+j%32 -> global col g*512 + u0 + (j%32).
    for (int i = 0; i < 768; ++i) {                 // 98304 elems / 128 thr
        int e = i * THREADS + tid;
        int k = e >> 7;                             // 0..767
        int j = e & 127;                            // local z column
        int col = (j >> 5) * U_ + u0 + (j & 31);
        float w = (k < D_) ? wk[k * G4U + col] : wr[(k - D_) * G4U + col];
        int kt = k >> 5, kk = k & 31;
        int nt = j >> 4, n = j & 15;
        // B fragment, this code's k order (as A): lanes 0-15 hold K 0-15, lanes 16-31 hold K 16-31
        int ln = n + ((kk >> 4) << 4);
        int ix = kk & 15;
        Bfrag[(nt * NKT + kt) * 512 + ln * 16 + ix] = (__bf16)w;
    }

    // ---- owner-stable per-thread state: unit u = u0+lane, rows m = wv + 4i ----
    const int u = u0 + lane;
    const float bi_ = bias[u];
    const float bf_ = bias[U_ + u];
    const float bc_ = bias[2 * U_ + u];
    const float bo_ = bias[3 * U_ + u];
    const float per_ = tg[u];
    const float shf_ = tg[U_ + u];
    const float ron_ = tg[2 * U_ + u];
    const float invper_ = 1.0f / per_;              // loop-invariant reciprocals:
    const float invron_ = 1.0f / ron_;              // kill per-step v_div sequences
    float hreg[4], creg[4];
    #pragma unroll
    for (int i = 0; i < 4; ++i) {
        int b = b0 + wv + 4 * i;
        hreg[i] = hbuf[b * U_ + u];                 // ping = h0 (init kernel ran first)
        creg[i] = c0[b * U_ + u];
    }
    __syncthreads();

    const int nt0 = wv * 2;                         // each wave owns 2 N-tiles
    const __bf16* bbase0 = Bfrag + (nt0 * NKT) * 512 + lane * 16;
    const __bf16* bbase1 = Bfrag + ((nt0 + 1) * NKT) * 512 + lane * 16;
    const __bf16* abase  = Afrag + lane * 16;

    for (int step = 0; step < T_; ++step) {
        const float* hin  = hbuf + (size_t)(step & 1) * (B_ * U_);
        float*       hout = hbuf + (size_t)((step + 1) & 1) * (B_ * U_);

        // ---- stage A = [x_t | h] (16 x 768) as bf16 A-fragments, K-pair packed ----
        // A fragment, this code's k order (as B): lane m holds K 0-15, lane m+16 holds K 16-31;
        // even/odd K land in adjacent halves of the same lane -> pack v2bf (b32 DS store).
        for (int i = 0; i < 48; ++i) {              // 6144 K-pairs / 128 thr
            int e = (i * THREADS + tid) * 2;
            int m = e / KTOT;
            int k = e - m * KTOT;                   // even; pair never straddles x|h (256 even)
            float2 v2 = (k < D_)
                ? *(const float2*)&x[((size_t)(b0 + m) * T_ + step) * D_ + k]
                : *(const float2*)&hin[(b0 + m) * U_ + (k - D_)];
            int kt = k >> 5, kk = k & 31;
            int ln = m + ((kk >> 4) << 4);
            int ix = kk & 15;                       // even
            v2bf p; p[0] = (__bf16)v2.x; p[1] = (__bf16)v2.y;
            *(v2bf*)(Afrag + kt * 512 + ln * 16 + ix) = p;
        }
        // prefetch next step's x tile into cache behind WMMA + barrier
        if (step + 1 < T_) {
            const float* nx = &x[((size_t)(b0 + (tid >> 3)) * T_ + (step + 1)) * D_
                                 + (tid & 7) * 32];
            __builtin_prefetch(nx, 0, 1);
            __builtin_prefetch(nx + 16, 0, 1);
        }
        __syncthreads();

        // ---- WMMA: z[16 x 128] = A[16 x 768] * Bslice[768 x 128], f32 acc ----
        v8f acc0 = {}; v8f acc1 = {};
        #pragma unroll 4
        for (int kt = 0; kt < NKT; ++kt) {
            v16bf a   = *(const v16bf*)(abase  + kt * 512);
            v16bf b0f = *(const v16bf*)(bbase0 + kt * 512);
            v16bf b1f = *(const v16bf*)(bbase1 + kt * 512);
            acc0 = __builtin_amdgcn_wmma_f32_16x16x32_bf16(
                false, a, false, b0f, (short)0, acc0, false, false);
            acc1 = __builtin_amdgcn_wmma_f32_16x16x32_bf16(
                false, a, false, b1f, (short)0, acc1, false, false);
        }
        // C/D layout: VGPR v, lane l: row = v + 8*(l>=16), col = ntile*16 + (l&15)
        // column-major z store: 8 contiguous floats per accumulator -> 2x ds_store_b128
        {
            int row0 = (lane >> 4) * 8;
            int cA = nt0 * 16 + (lane & 15);
            int cB = (nt0 + 1) * 16 + (lane & 15);
            *(v8f*)(zsh + cA * 16 + row0) = acc0;
            *(v8f*)(zsh + cB * 16 + row0) = acc1;
        }
        __syncthreads();

        // ---- gates + phased time-gate update (f32, register-resident h/c) ----
        float tnow = (float)(step + 1);
        float ts  = tnow - shf_;
        float phi = (ts - per_ * floorf(ts * invper_)) * invper_;
        float kk  = (phi <= 0.5f * ron_) ? (2.0f * phi * invron_)
                  : ((phi <= ron_) ? (2.0f - 2.0f * phi * invron_) : (ALPHA * phi));
        #pragma unroll
        for (int i = 0; i < 4; ++i) {
            int m = wv + 4 * i;
            int b = b0 + m;
            float zi = zsh[(lane)       * 16 + m] + bi_;
            float zf = zsh[(lane + 32)  * 16 + m] + bf_;
            float zc = zsh[(lane + 64)  * 16 + m] + bc_;
            float zo = zsh[(lane + 96)  * 16 + m] + bo_;
            float ig = fminf(fmaxf(0.2f * zi + 0.5f, 0.0f), 1.0f);
            float fg = fminf(fmaxf(0.2f * zf + 0.5f, 0.0f), 1.0f);
            float og = fminf(fmaxf(0.2f * zo + 0.5f, 0.0f), 1.0f);
            float chat = fg * creg[i] + ig * htanh(zc);
            float hhat = og * htanh(chat);
            float cnew = kk * chat + (1.0f - kk) * creg[i];
            float hnew = kk * hhat + (1.0f - kk) * hreg[i];
            creg[i] = cnew;
            hreg[i] = hnew;
            hout[b * U_ + u] = hnew;
            out[((size_t)b * T_ + step) * U_ + u] = hnew;
        }

        // ---- device-scope barrier: publish hout before anyone reads it ----
        __threadfence();
        __syncthreads();
        if (tid == 0) {
            __hip_atomic_fetch_add(bar, 1, __ATOMIC_ACQ_REL, __HIP_MEMORY_SCOPE_AGENT);
            int tgt = NWG * (step + 1);
            while (__hip_atomic_load(bar, __ATOMIC_ACQUIRE, __HIP_MEMORY_SCOPE_AGENT) < tgt) {
                __builtin_amdgcn_s_sleep(2);
            }
        }
        __syncthreads();
    }
}

extern "C" void kernel_launch(void* const* d_in, const int* in_sizes, int n_in,
                              void* d_out, int out_size, void* d_ws, size_t ws_size,
                              hipStream_t stream) {
    const float* x   = (const float*)d_in[0];
    const float* h0  = (const float*)d_in[1];
    const float* c0  = (const float*)d_in[2];
    // d_in[3] = t0 (zeros; t = step+1 reproduces the reference's t accumulation)
    const float* wk  = (const float*)d_in[4];
    const float* wr  = (const float*)d_in[5];
    const float* bias= (const float*)d_in[6];
    const float* tg  = (const float*)d_in[7];
    float* out  = (float*)d_out;
    float* hbuf = (float*)d_ws;                               // 2 * 128*512 f32 = 512 KB
    int*   bar  = (int*)((char*)d_ws + 2 * B_ * U_ * sizeof(float));

    plstm_init<<<(B_ * U_ + 255) / 256, 256, 0, stream>>>(h0, hbuf, bar);
    plstm_persistent<<<NWG, THREADS, SMEM_BYTES, stream>>>(
        x, c0, wk, wr, bias, tg, out, hbuf, bar);
}
